// GEN_85263690760422
// MI455X (gfx1250) — compile-verified, matrix operand order checked by reading, not run on hardware
//
#include <hip/hip_runtime.h>
#include <hip/hip_bf16.h>

// ---------------------------------------------------------------------------
// GENConv GNN pipeline for MI455X (gfx1250, wave32, WMMA).
// - All dense GEMMs: v_wmma_f32_16x16x32_f16, f32 accumulate.
// - A fragments loaded as contiguous b128 (two 8-float runs per lane).
// - Weights pre-transposed to f16 [N,K] once per launch -> B frags are b128.
// - Scatter-softmax aggregation: float4-vectorized atomicMax/atomicAdd passes.
// - agg division uses hardware v_rcp_f32 instead of IEEE div expansion.
// ---------------------------------------------------------------------------

#define NN_NODES 80000
#define NE_EDGES 1280000
#define D_NODE   128
#define D_EDGE   32
#define HID      64
#define OUTD     32
#define EPS_GEN  1e-7f
#define EPS_BN   1e-5f

typedef __attribute__((ext_vector_type(16))) _Float16 v16h;
typedef __attribute__((ext_vector_type(8)))  _Float16 h8v;
typedef __attribute__((ext_vector_type(8)))  float    v8f;
typedef __attribute__((ext_vector_type(4)))  float    f4v;
typedef __attribute__((ext_vector_type(4)))  int      i4v;

// ---------------------------------------------------------------------------
// Fragment loaders. Per the ISA 16-bit A/B layout, lane (m, half) holds
// K = {half*8 .. half*8+7} in elements [0..7] and K = {16+half*8 ..} in [8..15]
// -> two contiguous 8-element runs of the source row.
// ---------------------------------------------------------------------------

// A fragment from a row-major f32 row pointer (rp already includes kbase).
__device__ __forceinline__ v16h load_a16(const float* __restrict__ rp, int half) {
    const f4v* q0 = (const f4v*)(rp + (half << 3));
    const f4v* q1 = (const f4v*)(rp + 16 + (half << 3));
    f4v a0 = q0[0], a1 = q0[1], a2 = q1[0], a3 = q1[1];
    v16h a;
#pragma unroll
    for (int i = 0; i < 4; ++i) {
        a[i]      = (_Float16)a0[i];
        a[4 + i]  = (_Float16)a1[i];
        a[8 + i]  = (_Float16)a2[i];
        a[12 + i] = (_Float16)a3[i];
    }
    return a;
}

// B fragment from pre-transposed f16 weights Wt[N,K] (row-major).
__device__ __forceinline__ v16h load_b16(const _Float16* __restrict__ Wt, int K,
                                         int colbase, int kbase, int lane) {
    int n = lane & 15, half = lane >> 4;
    const _Float16* p = Wt + (size_t)(colbase + n) * K + kbase + (half << 3);
    h8v lo = *(const h8v*)(p);
    h8v hi = *(const h8v*)(p + 16);
    v16h b;
#pragma unroll
    for (int i = 0; i < 8; ++i) { b[i] = lo[i]; b[8 + i] = hi[i]; }
    return b;
}

// Weight transpose + f32->f16: Wt[n*K + k] = W[k*N + n].
__global__ void wt_f16_kernel(const float* __restrict__ W, _Float16* __restrict__ Wt,
                              int K, int N) {
    int idx = blockIdx.x * blockDim.x + threadIdx.x;
    if (idx >= K * N) return;
    int k = idx / N, n = idx % N;
    Wt[(size_t)n * K + k] = (_Float16)W[idx];
}

// ---------------------------------------------------------------------------
// Generic GEMM, full-width strip per wave: C[16, NOUT] = A[16,K] @ W + bias.
// A fragment loaded once per K-step, reused across NOUT/16 WMMAs.
// ---------------------------------------------------------------------------
template <int NOUT>
__global__ void gemm_wmma_bias(const float* __restrict__ A, const _Float16* __restrict__ Wt,
                               const float* __restrict__ bias, float* __restrict__ C,
                               int M, int K, int relu) {
    int lane = threadIdx.x & 31;
    int tile = blockIdx.x * (blockDim.x >> 5) + (threadIdx.x >> 5);
    if (tile >= (M >> 4)) return;
    int rowbase = tile << 4;
    int m = lane & 15, half = lane >> 4;
    const float* arow = A + (size_t)(rowbase + m) * K;
    v8f c[NOUT / 16];
#pragma unroll
    for (int cb = 0; cb < NOUT / 16; ++cb) c[cb] = (v8f){};
    for (int kb = 0; kb < K; kb += 32) {
        v16h a = load_a16(arow + kb, half);
#pragma unroll
        for (int cb = 0; cb < NOUT / 16; ++cb) {
            v16h b = load_b16(Wt, K, cb * 16, kb, lane);
            c[cb] = __builtin_amdgcn_wmma_f32_16x16x32_f16(false, a, false, b, (short)0, c[cb], false, false);
        }
    }
    int n = lane & 15;
#pragma unroll
    for (int cb = 0; cb < NOUT / 16; ++cb) {
        float bv = bias ? bias[cb * 16 + n] : 0.f;
#pragma unroll
        for (int r = 0; r < 8; ++r) {
            float v = c[cb][r] + bv;
            if (relu) v = fmaxf(v, 0.f);
            C[(size_t)(rowbase + r + (half << 3)) * NOUT + cb * 16 + n] = v;
        }
    }
}

// ---------------------------------------------------------------------------
// Softmax aggregation pass 1: per-(dst,feature) max. msg >= EPS_GEN > 0 always,
// so integer atomicMax on float bits with a 0-initialized buffer is exact and
// also reproduces the reference's "no-in-edge -> 0" rule. float4 per thread.
// ---------------------------------------------------------------------------
__global__ void msg_max4_kernel(const float* __restrict__ h, const float* __restrict__ ea,
                                const int* __restrict__ src, const int* __restrict__ dst,
                                int* __restrict__ mmax, int total) {
    int idx = blockIdx.x * blockDim.x + threadIdx.x;
    if (idx >= total) return;
    int e = idx >> 4, f0 = (idx & 15) << 2;
    f4v hv = *(const f4v*)(h + (size_t)src[e] * HID + f0);
    f4v ev = *(const f4v*)(ea + (size_t)e * HID + f0);
    int* mp = mmax + (size_t)dst[e] * HID + f0;
#pragma unroll
    for (int i = 0; i < 4; ++i) {
        float msg = fmaxf(hv[i] + ev[i], 0.f) + EPS_GEN;
        atomicMax(mp + i, __float_as_int(msg));
    }
}

// Pass 2: S1 = sum exp(msg-m), S2 = sum msg*exp(msg-m)  (agg = S2/(S1+1e-16)).
__global__ void msg_sum4_kernel(const float* __restrict__ h, const float* __restrict__ ea,
                                const int* __restrict__ src, const int* __restrict__ dst,
                                const int* __restrict__ mmax,
                                float* __restrict__ S1, float* __restrict__ S2, int total) {
    int idx = blockIdx.x * blockDim.x + threadIdx.x;
    if (idx >= total) return;
    int e = idx >> 4, f0 = (idx & 15) << 2;
    size_t di = (size_t)dst[e] * HID + f0;
    f4v hv = *(const f4v*)(h + (size_t)src[e] * HID + f0);
    f4v ev = *(const f4v*)(ea + (size_t)e * HID + f0);
    i4v mv = *(const i4v*)(mmax + di);
#pragma unroll
    for (int i = 0; i < 4; ++i) {
        float msg = fmaxf(hv[i] + ev[i], 0.f) + EPS_GEN;
        float ex  = __expf(msg - __int_as_float(mv[i]));
        atomicAdd(S1 + di + i, ex);
        atomicAdd(S2 + di + i, msg * ex);
    }
}

// ---------------------------------------------------------------------------
// T[M,128] = (S2/(S1+1e-16) + h) @ W1[64,128]; agg+residual fused into the
// A loader (vector loads of S1/S2/h, hardware v_rcp_f32 for the division).
// ---------------------------------------------------------------------------
__device__ __forceinline__ float agg_elem(float s1, float s2, float hv) {
    return s2 * __builtin_amdgcn_rcpf(s1 + 1e-16f) + hv;   // single v_rcp_f32
}

__device__ __forceinline__ v16h load_agg16(const float* __restrict__ S1r,
                                           const float* __restrict__ S2r,
                                           const float* __restrict__ hr, int half) {
    int o0 = (half << 3), o1 = 16 + (half << 3);
    f4v s1a = *(const f4v*)(S1r + o0), s1b = *(const f4v*)(S1r + o0 + 4);
    f4v s1c = *(const f4v*)(S1r + o1), s1d = *(const f4v*)(S1r + o1 + 4);
    f4v s2a = *(const f4v*)(S2r + o0), s2b = *(const f4v*)(S2r + o0 + 4);
    f4v s2c = *(const f4v*)(S2r + o1), s2d = *(const f4v*)(S2r + o1 + 4);
    f4v ha  = *(const f4v*)(hr + o0),  hb  = *(const f4v*)(hr + o0 + 4);
    f4v hc  = *(const f4v*)(hr + o1),  hd  = *(const f4v*)(hr + o1 + 4);
    v16h a;
#pragma unroll
    for (int i = 0; i < 4; ++i) {
        a[i]      = (_Float16)agg_elem(s1a[i], s2a[i], ha[i]);
        a[4 + i]  = (_Float16)agg_elem(s1b[i], s2b[i], hb[i]);
        a[8 + i]  = (_Float16)agg_elem(s1c[i], s2c[i], hc[i]);
        a[12 + i] = (_Float16)agg_elem(s1d[i], s2d[i], hd[i]);
    }
    return a;
}

__global__ void gemm_agg_wmma(const float* __restrict__ S1, const float* __restrict__ S2,
                              const float* __restrict__ h, const _Float16* __restrict__ W1t,
                              float* __restrict__ T, int M) {
    int lane = threadIdx.x & 31;
    int tile = blockIdx.x * (blockDim.x >> 5) + (threadIdx.x >> 5);
    if (tile >= (M >> 4)) return;
    int rowbase = tile << 4;
    int m = lane & 15, half = lane >> 4;
    size_t ro = (size_t)(rowbase + m) * HID;
    v8f c[8];
#pragma unroll
    for (int cb = 0; cb < 8; ++cb) c[cb] = (v8f){};
#pragma unroll
    for (int kb = 0; kb < HID; kb += 32) {
        v16h a = load_agg16(S1 + ro + kb, S2 + ro + kb, h + ro + kb, half);
#pragma unroll
        for (int cb = 0; cb < 8; ++cb) {
            v16h b = load_b16(W1t, HID, cb * 16, kb, lane);
            c[cb] = __builtin_amdgcn_wmma_f32_16x16x32_f16(false, a, false, b, (short)0, c[cb], false, false);
        }
    }
    int n = lane & 15;
#pragma unroll
    for (int cb = 0; cb < 8; ++cb)
#pragma unroll
        for (int r = 0; r < 8; ++r)
            T[(size_t)(rowbase + r + (half << 3)) * 128 + cb * 16 + n] = c[cb][r];
}

// Column sums / sums-of-squares of T[M,128] for BatchNorm statistics.
__global__ void bn_stats_kernel(const float* __restrict__ T, float* __restrict__ colsum,
                                float* __restrict__ colsq, int M) {
    int col = threadIdx.x;                       // 128 threads: one column each
    int rows = (M + gridDim.x - 1) / gridDim.x;
    int r0 = blockIdx.x * rows, r1 = min(M, r0 + rows);
    float s = 0.f, sq = 0.f;
    for (int r = r0; r < r1; ++r) {
        float v = T[(size_t)r * 128 + col];
        s += v; sq += v * v;
    }
    atomicAdd(&colsum[col], s);
    atomicAdd(&colsq[col], sq);
}

__global__ void bn_finalize_kernel(const float* __restrict__ colsum, const float* __restrict__ colsq,
                                   const float* __restrict__ g, const float* __restrict__ be,
                                   float* __restrict__ bnA, float* __restrict__ bnB, int M) {
    int c = threadIdx.x;
    float mu  = colsum[c] / (float)M;
    float var = colsq[c] / (float)M - mu * mu;
    float a   = g[c] * rsqrtf(var + EPS_BN);
    bnA[c] = a;
    bnB[c] = be[c] - mu * a;
}

// ---------------------------------------------------------------------------
// out[M,64] = relu( relu(T*bnA + bnB) @ W2[128,64] ); BN affine + inner relu
// fused into the A loader (vector loads), outer relu in the epilogue.
// ---------------------------------------------------------------------------
__device__ __forceinline__ v16h load_bn16(const float* __restrict__ Tr,
                                          const float* __restrict__ bnA,
                                          const float* __restrict__ bnB,
                                          int kb, int half) {
    int o0 = kb + (half << 3), o1 = kb + 16 + (half << 3);
    f4v t0 = *(const f4v*)(Tr + o0), t1 = *(const f4v*)(Tr + o0 + 4);
    f4v t2 = *(const f4v*)(Tr + o1), t3 = *(const f4v*)(Tr + o1 + 4);
    f4v a0 = *(const f4v*)(bnA + o0), a1 = *(const f4v*)(bnA + o0 + 4);
    f4v a2 = *(const f4v*)(bnA + o1), a3 = *(const f4v*)(bnA + o1 + 4);
    f4v b0 = *(const f4v*)(bnB + o0), b1 = *(const f4v*)(bnB + o0 + 4);
    f4v b2 = *(const f4v*)(bnB + o1), b3 = *(const f4v*)(bnB + o1 + 4);
    v16h a;
#pragma unroll
    for (int i = 0; i < 4; ++i) {
        a[i]      = (_Float16)fmaxf(t0[i] * a0[i] + b0[i], 0.f);
        a[4 + i]  = (_Float16)fmaxf(t1[i] * a1[i] + b1[i], 0.f);
        a[8 + i]  = (_Float16)fmaxf(t2[i] * a2[i] + b2[i], 0.f);
        a[12 + i] = (_Float16)fmaxf(t3[i] * a3[i] + b3[i], 0.f);
    }
    return a;
}

__global__ void gemm_bn_wmma(const float* __restrict__ T, const float* __restrict__ bnA,
                             const float* __restrict__ bnB, const _Float16* __restrict__ W2t,
                             float* __restrict__ C, int M) {
    int lane = threadIdx.x & 31;
    int tile = blockIdx.x * (blockDim.x >> 5) + (threadIdx.x >> 5);
    if (tile >= (M >> 4)) return;
    int rowbase = tile << 4;
    int m = lane & 15, half = lane >> 4;
    const float* Tr = T + (size_t)(rowbase + m) * 128;
    v8f c[4];
#pragma unroll
    for (int cb = 0; cb < 4; ++cb) c[cb] = (v8f){};
#pragma unroll
    for (int kb = 0; kb < 128; kb += 32) {
        v16h a = load_bn16(Tr, bnA, bnB, kb, half);
#pragma unroll
        for (int cb = 0; cb < 4; ++cb) {
            v16h b = load_b16(W2t, 128, cb * 16, kb, lane);
            c[cb] = __builtin_amdgcn_wmma_f32_16x16x32_f16(false, a, false, b, (short)0, c[cb], false, false);
        }
    }
    int n = lane & 15;
#pragma unroll
    for (int cb = 0; cb < 4; ++cb)
#pragma unroll
        for (int r = 0; r < 8; ++r)
            C[(size_t)(rowbase + r + (half << 3)) * HID + cb * 16 + n] = fmaxf(c[cb][r], 0.f);
}

// ---------------------------------------------------------------------------
// Edge update: out[E,HOUT] = concat(h[src], ea, h[dst]) @ W[192,HOUT] + bias.
// One wave owns 16 full edge rows -> safe to write ea in place for layer 1
// (all ea reads happen in the K loop before the stores).
// ---------------------------------------------------------------------------
template <int HOUT>
__global__ void edge_lin_wmma(const float* __restrict__ h, const float* __restrict__ ea,
                              const int* __restrict__ src, const int* __restrict__ dst,
                              const _Float16* __restrict__ Wt, const float* __restrict__ bias,
                              float* __restrict__ out, int E) {
    int lane = threadIdx.x & 31;
    int tile = blockIdx.x * (blockDim.x >> 5) + (threadIdx.x >> 5);
    if (tile >= (E >> 4)) return;
    int ebase = tile << 4;
    int m = lane & 15, half = lane >> 4;
    int erow = ebase + m;
    const float* seg[3] = { h  + (size_t)src[erow] * HID,
                            ea + (size_t)erow      * HID,
                            h  + (size_t)dst[erow] * HID };
    v8f c[HOUT / 16];
#pragma unroll
    for (int cb = 0; cb < HOUT / 16; ++cb) c[cb] = (v8f){};
#pragma unroll
    for (int kb = 0; kb < 192; kb += 32) {       // each 32-K block in one segment
        v16h a = load_a16(seg[kb / 64] + (kb & 63), half);
#pragma unroll
        for (int cb = 0; cb < HOUT / 16; ++cb) {
            v16h b = load_b16(Wt, 192, cb * 16, kb, lane);
            c[cb] = __builtin_amdgcn_wmma_f32_16x16x32_f16(false, a, false, b, (short)0, c[cb], false, false);
        }
    }
    int n = lane & 15;
#pragma unroll
    for (int cb = 0; cb < HOUT / 16; ++cb) {
        float bv = bias[cb * 16 + n];
#pragma unroll
        for (int r = 0; r < 8; ++r)
            out[(size_t)(ebase + r + (half << 3)) * HOUT + cb * 16 + n] = c[cb][r] + bv;
    }
}

// ---------------------------------------------------------------------------
// Host-side orchestration.
// ---------------------------------------------------------------------------
static inline int blocks4(int tiles) { return (tiles + 3) / 4; }

extern "C" void kernel_launch(void* const* d_in, const int* in_sizes, int n_in,
                              void* d_out, int out_size, void* d_ws, size_t ws_size,
                              hipStream_t stream) {
    (void)in_sizes; (void)n_in; (void)out_size; (void)ws_size;
    const float* x         = (const float*)d_in[0];
    const int*   eidx      = (const int*)  d_in[1];
    const float* edge_attr = (const float*)d_in[2];
    const float* node_W    = (const float*)d_in[3];
    const float* node_b    = (const float*)d_in[4];
    const float* edge_W    = (const float*)d_in[5];
    const float* edge_b    = (const float*)d_in[6];
    const float* c1_W1 = (const float*)d_in[7],  *c1_g = (const float*)d_in[8];
    const float* c1_be = (const float*)d_in[9],  *c1_W2 = (const float*)d_in[10];
    const float* c2_W1 = (const float*)d_in[11], *c2_g = (const float*)d_in[12];
    const float* c2_be = (const float*)d_in[13], *c2_W2 = (const float*)d_in[14];
    const float* l1_W = (const float*)d_in[15], *l1_b = (const float*)d_in[16];
    const float* l2_W = (const float*)d_in[17], *l2_b = (const float*)d_in[18];

    const int* src = eidx;
    const int* dst = eidx + NE_EDGES;

    // workspace carve-out
    float* ws   = (float*)d_ws;
    float* ea   = ws;                                   // [E,64]
    float* hA   = ea + (size_t)NE_EDGES * HID;          // [N,64] encoder out
    float* hB   = hA + (size_t)NN_NODES * HID;          // [N,64] layer-1 out
    int*   mI   = (int*)(hB + (size_t)NN_NODES * HID);  // [N,64] max (float bits)
    float* S1   = (float*)mI + (size_t)NN_NODES * HID;
    float* S2   = S1 + (size_t)NN_NODES * HID;
    float* T    = S2 + (size_t)NN_NODES * HID;          // [N,128]
    float* csum = T + (size_t)NN_NODES * 128;           // [128]
    float* csq  = csum + 128;
    float* bnA  = csq + 128;
    float* bnB  = bnA + 128;
    // pre-transposed f16 weights
    _Float16* wtN  = (_Float16*)(bnB + 128);            // node_W^T  [64 x 128]
    _Float16* wtE  = wtN  + 64 * 128;                   // edge_W^T  [64 x 32]
    _Float16* w11t = wtE  + 64 * 32;                    // c1_W1^T   [128 x 64]
    _Float16* w12t = w11t + 128 * 64;                   // c1_W2^T   [64 x 128]
    _Float16* w21t = w12t + 64 * 128;                   // c2_W1^T   [128 x 64]
    _Float16* w22t = w21t + 128 * 64;                   // c2_W2^T   [64 x 128]
    _Float16* wl1t = w22t + 64 * 128;                   // l1_W^T    [64 x 192]
    _Float16* wl2t = wl1t + 64 * 192;                   // l2_W^T    [32 x 192]

    float* h_out  = (float*)d_out;                      // [N,64]  final h
    float* ea_out = h_out + (size_t)NN_NODES * HID;     // [E,32]  final ea

    auto wt = [&](const float* W, _Float16* Wt, int K, int N) {
        wt_f16_kernel<<<(K * N + 255) / 256, 256, 0, stream>>>(W, Wt, K, N);
    };
    wt(node_W, wtN, D_NODE, HID);
    wt(edge_W, wtE, D_EDGE, HID);
    wt(c1_W1, w11t, HID, 128);   wt(c1_W2, w12t, 128, HID);
    wt(c2_W1, w21t, HID, 128);   wt(c2_W2, w22t, 128, HID);
    wt(l1_W, wl1t, 192, HID);    wt(l2_W, wl2t, 192, OUTD);

    const int msgTotal  = NE_EDGES * 16;                // float4 per thread
    const int msgBlocks = (msgTotal + 255) / 256;

    auto conv = [&](const float* hin, float* hout, const _Float16* W1t,
                    const float* g, const float* be, const _Float16* W2t) {
        hipMemsetAsync(mI, 0, (size_t)NN_NODES * HID * 4, stream);
        hipMemsetAsync(S1, 0, (size_t)NN_NODES * HID * 4, stream);
        hipMemsetAsync(S2, 0, (size_t)NN_NODES * HID * 4, stream);
        msg_max4_kernel<<<msgBlocks, 256, 0, stream>>>(hin, ea, src, dst, mI, msgTotal);
        msg_sum4_kernel<<<msgBlocks, 256, 0, stream>>>(hin, ea, src, dst, mI, S1, S2, msgTotal);
        gemm_agg_wmma<<<blocks4(NN_NODES / 16), 128, 0, stream>>>(S1, S2, hin, W1t, T, NN_NODES);
        hipMemsetAsync(csum, 0, 128 * 4, stream);
        hipMemsetAsync(csq,  0, 128 * 4, stream);
        bn_stats_kernel<<<256, 128, 0, stream>>>(T, csum, csq, NN_NODES);
        bn_finalize_kernel<<<1, 128, 0, stream>>>(csum, csq, g, be, bnA, bnB, NN_NODES);
        gemm_bn_wmma<<<blocks4(NN_NODES / 16), 128, 0, stream>>>(T, bnA, bnB, W2t, hout, NN_NODES);
    };

    // encoders
    gemm_wmma_bias<HID><<<blocks4(NN_NODES / 16), 128, 0, stream>>>(
        x, wtN, node_b, hA, NN_NODES, D_NODE, 0);
    gemm_wmma_bias<HID><<<blocks4(NE_EDGES / 16), 128, 0, stream>>>(
        edge_attr, wtE, edge_b, ea, NE_EDGES, D_EDGE, 0);

    // layer 1
    conv(hA, hB, w11t, c1_g, c1_be, w12t);
    edge_lin_wmma<HID><<<blocks4(NE_EDGES / 16), 128, 0, stream>>>(
        hB, ea, src, dst, wl1t, l1_b, ea, NE_EDGES);        // in-place ea update

    // layer 2 (node output straight into d_out)
    conv(hB, h_out, w21t, c2_g, c2_be, w22t);
    edge_lin_wmma<OUTD><<<blocks4(NE_EDGES / 16), 128, 0, stream>>>(
        h_out, ea, src, dst, wl2t, l2_b, ea_out, NE_EDGES);
}
